// MultiHeadAttention_14405320311016
// MI455X (gfx1250) — matrix operand order repaired and checked by reading, not compiled, not run on hardware
//
#include <hip/hip_runtime.h>

// ---------------------------------------------------------------------------
// MHA forward for MI455X (gfx1250, wave32, WMMA bf16 w/ f32 accumulate)
// B=2, S=2048, D=1024, H=16, DK=64
// ---------------------------------------------------------------------------

typedef unsigned short u16;
typedef __attribute__((ext_vector_type(16))) __bf16 v16bf;
typedef __attribute__((ext_vector_type(16))) u16    us16;
typedef __attribute__((ext_vector_type(8)))  float  v8f;
typedef __attribute__((ext_vector_type(4)))  float  f4;

constexpr int Bc = 2, Sc = 2048, Dc = 1024, Hc = 16, DKc = 64;
constexpr int Rc = Bc * Sc;   // 4096 rows for the projection GEMMs

__device__ __forceinline__ u16 f2bf(float f) {
  unsigned u = __builtin_bit_cast(unsigned, f);
  u += 0x7fffu + ((u >> 16) & 1u);          // round-to-nearest-even
  return (u16)(u >> 16);
}

__device__ __forceinline__ v8f wmma_bf16(v16bf a, v16bf b, v8f c) {
  return __builtin_amdgcn_wmma_f32_16x16x32_bf16(
      /*neg_a=*/false, a, /*neg_b=*/false, b,
      /*c_mod=*/(short)0, c, /*reuse_a=*/false, /*reuse_b=*/false);
}

// A-matrix (16x32 bf16): lane holds row, two 8-elem K chunks at half*8 and
// 16+half*8 (ISA 16-bit A layout; elems 0..7 -> VGPR0..3, 8..15 -> VGPR4..7).
__device__ __forceinline__ v16bf load_a_bf(const u16* rowptr, int half) {
  us16 r;
  const u16* p0 = rowptr + half * 8;
  const u16* p1 = rowptr + 16 + half * 8;
#pragma unroll
  for (int i = 0; i < 8; i++) r[i] = p0[i];
#pragma unroll
  for (int i = 0; i < 8; i++) r[8 + i] = p1[i];
  return __builtin_bit_cast(v16bf, r);
}

// B-matrix (32x16 bf16): lane holds one column, K chunks at half*8 and 16+half*8 (p = base + half*8).
__device__ __forceinline__ v16bf load_b_bf(const u16* p) {
  us16 r;
#pragma unroll
  for (int i = 0; i < 8; i++) r[i] = p[i];
#pragma unroll
  for (int i = 0; i < 8; i++) r[8 + i] = p[16 + i];
  return __builtin_bit_cast(v16bf, r);
}

// ---------------------------------------------------------------------------
// fp32 -> bf16 bulk convert (bandwidth-bound pre-pass).
// ---------------------------------------------------------------------------
__global__ __launch_bounds__(256) void cvt_bf16_kernel(const float* __restrict__ in,
                                                       u16* __restrict__ out, int n) {
  const int i = (blockIdx.x * blockDim.x + threadIdx.x) * 4;
  if (i + 3 < n) {
    f4 v = *(const f4*)(in + i);
    out[i + 0] = f2bf(v[0]);
    out[i + 1] = f2bf(v[1]);
    out[i + 2] = f2bf(v[2]);
    out[i + 3] = f2bf(v[3]);
  }
}

// ---------------------------------------------------------------------------
// Projection GEMM: C[R, D] = X[R, D] @ W^T + bias, one wave -> 16x64 slab.
// X and W are pre-converted bf16.
// OUTMODE 0: bf16 out, layout [B,H,S,DK]         (Q and K)
// OUTMODE 1: bf16 out, layout [B,H,DK,S]         (V transposed)
// OUTMODE 2: fp32 out, layout [R, D]             (final output)
// ---------------------------------------------------------------------------
template <int OUTMODE>
__global__ __launch_bounds__(256) void proj_kernel(const u16* __restrict__ X,
                                                   const u16* __restrict__ W,
                                                   const float* __restrict__ bias,
                                                   void* __restrict__ Out,
                                                   float scale) {
  const int lane = threadIdx.x & 31;
  const int wid  = blockIdx.x * (blockDim.x >> 5) + (threadIdx.x >> 5);
  const int TN = Dc / 64;               // 16 column groups of 64
  const int mt = wid / TN;
  const int ng = wid % TN;
  if (mt >= Rc / 16) return;
  const int m0 = mt * 16;
  const int n0 = ng * 64;
  const int half = lane >> 4;
  const int lr   = lane & 15;
  const int arow = m0 + lr;

  v8f acc[4] = {};
  for (int k0 = 0; k0 < Dc; k0 += 32) {
    v16bf a = load_a_bf(X + (size_t)arow * Dc + k0, half);
#pragma unroll
    for (int t = 0; t < 4; t++) {
      const u16* wp = W + (size_t)(n0 + t * 16 + lr) * Dc + k0 + half * 8;
      acc[t] = wmma_bf16(a, load_b_bf(wp), acc[t]);
    }
  }

#pragma unroll
  for (int t = 0; t < 4; t++) {
    const int n  = n0 + t * 16 + lr;
    const float bv = bias[n];
#pragma unroll
    for (int v = 0; v < 8; v++) {
      const int gm  = m0 + v + half * 8;       // global row in [0, R)
      const float val = (acc[t][v] + bv) * scale;
      if constexpr (OUTMODE == 0) {
        const int bb = gm >> 11, s = gm & (Sc - 1);
        const int h = n >> 6, dk = n & 63;
        ((u16*)Out)[(((size_t)(bb * Hc + h) * Sc + s) * DKc) + dk] = f2bf(val);
      } else if constexpr (OUTMODE == 1) {
        const int bb = gm >> 11, s = gm & (Sc - 1);
        const int h = n >> 6, dk = n & 63;
        ((u16*)Out)[(((size_t)(bb * Hc + h) * DKc + dk) * Sc) + s] = f2bf(val);
      } else {
        ((float*)Out)[(size_t)gm * Dc + n] = val;
      }
    }
  }
}

// ---------------------------------------------------------------------------
// Flash-style causal attention, transposed softmax.
// One wave = 16-row query tile of one (b,h); 64-key blocks:
//   8 score WMMAs -> fp32 scores via LDS into row-per-lane layout ->
//   in-lane max/exp/sum + single shfl_xor(16) -> P already in A layout ->
//   8 P@V WMMAs.  Running m/l are one scalar per lane (row = lane&15).
// ---------------------------------------------------------------------------
__global__ __launch_bounds__(256) void attn_kernel(const u16* __restrict__ Qb,
                                                   const u16* __restrict__ Kb,
                                                   const u16* __restrict__ Vt,
                                                   u16* __restrict__ Ob) {
  __shared__ float sT[8][16 * 64];      // per-wave 16x64 fp32 score tile (4 KB)
  __shared__ float sR[8][16];           // per-wave row-stat broadcast line
  const int lane  = threadIdx.x & 31;
  const int wslot = threadIdx.x >> 5;
  const int wid   = blockIdx.x * 8 + wslot;
  const int QT = Sc / 16;               // 128 query tiles per head
  const int qt = wid % QT;
  const int bh = wid / QT;              // 0 .. B*H-1
  if (bh >= Bc * Hc) return;
  const int q0 = qt * 16;
  const int half = lane >> 4;
  const int lr   = lane & 15;
  float* T  = sT[wslot];
  float* Rb = sR[wslot];

  const u16* qbase = Qb + (size_t)bh * Sc * DKc;
  const u16* kbase = Kb + (size_t)bh * Sc * DKc;
  const u16* vbase = Vt + (size_t)bh * DKc * Sc;

  // Q tile 16x64 as two A operands (features 0..31 and 32..63)
  const v16bf qa0 = load_a_bf(qbase + (size_t)(q0 + lr) * DKc, half);
  const v16bf qa1 = load_a_bf(qbase + (size_t)(q0 + lr) * DKc + 32, half);

  v8f o[4] = {};
  float m = -3.0e38f, l = 0.0f;         // running stats for row q0+lr
  const int qrow = q0 + lr;

  for (int j0 = 0; j0 <= q0; j0 += 64) {
    // ---- scores: S = Q*K^T, four 16x16 N-tiles (keys j0+16t .. +15)
    v8f s[4];
#pragma unroll
    for (int t = 0; t < 4; t++) {
      const u16* kp = kbase + (size_t)(j0 + t * 16 + lr) * DKc + half * 8;
      v8f z = {};
      z = wmma_bf16(qa0, load_b_bf(kp), z);
      s[t] = wmma_bf16(qa1, load_b_bf(kp + 32), z);
    }

    // ---- C-layout -> LDS (fp32), then read back row-per-lane (A layout)
#pragma unroll
    for (int t = 0; t < 4; t++)
#pragma unroll
      for (int v = 0; v < 8; v++)
        T[(v + half * 8) * 64 + t * 16 + lr] = s[t][v];
    asm volatile("s_wait_dscnt 0" ::: "memory");

    float f[32];                         // this half-lane's 32 of 64 keys
#pragma unroll
    for (int c = 0; c < 4; c++)
#pragma unroll
      for (int i = 0; i < 8; i++)
        f[c * 8 + i] = T[lr * 64 + c * 16 + half * 8 + i];

    // ---- causal mask (only diagonal/overshoot blocks) + row max
    float mx = -3.0e38f;
    if (j0 + 63 > q0) {
#pragma unroll
      for (int c = 0; c < 4; c++)
#pragma unroll
        for (int i = 0; i < 8; i++) {
          const int key = j0 + c * 16 + half * 8 + i;
          if (key > qrow) f[c * 8 + i] = -3.0e38f;
          mx = fmaxf(mx, f[c * 8 + i]);
        }
    } else {
#pragma unroll
      for (int k = 0; k < 32; k++) mx = fmaxf(mx, f[k]);
    }
    mx = fmaxf(mx, __shfl_xor(mx, 16));  // combine the two half-rows

    const float mn   = fmaxf(m, mx);
    const float corr = __expf(m - mn);
    m = mn;
    float rs = 0.0f;
#pragma unroll
    for (int k = 0; k < 32; k++) {
      f[k] = __expf(f[k] - mn);
      rs += f[k];
    }
    rs += __shfl_xor(rs, 16);
    l = l * corr + rs;

    // ---- pack P (already row-per-lane == A layout)
    us16 r0, r1;
#pragma unroll
    for (int i = 0; i < 8; i++) {
      r0[i]     = f2bf(f[i]);
      r0[8 + i] = f2bf(f[8 + i]);
      r1[i]     = f2bf(f[16 + i]);
      r1[8 + i] = f2bf(f[24 + i]);
    }
    const v16bf pa0 = __builtin_bit_cast(v16bf, r0);
    const v16bf pa1 = __builtin_bit_cast(v16bf, r1);

    // ---- broadcast corr back to C-layout rows and rescale O
    Rb[lr] = corr;                       // both halves write identical value
    asm volatile("s_wait_dscnt 0" ::: "memory");
    float cf[8];
#pragma unroll
    for (int v = 0; v < 8; v++) cf[v] = Rb[half * 8 + v];
#pragma unroll
    for (int dt = 0; dt < 4; dt++)
#pragma unroll
      for (int v = 0; v < 8; v++) o[dt][v] *= cf[v];

    // ---- O += P * V ; V^T layout makes B-operands contiguous per lane
#pragma unroll
    for (int dt = 0; dt < 4; dt++) {
      const u16* vp = vbase + (size_t)(dt * 16 + lr) * Sc + j0;
      o[dt] = wmma_bf16(pa0, load_b_bf(vp + half * 8), o[dt]);
      o[dt] = wmma_bf16(pa1, load_b_bf(vp + 32 + half * 8), o[dt]);
    }
  }

  // ---- normalize (1/l broadcast) and store bf16 into [B, S, D]
  Rb[lr] = l;
  asm volatile("s_wait_dscnt 0" ::: "memory");
  float lf[8];
#pragma unroll
  for (int v = 0; v < 8; v++) lf[v] = Rb[half * 8 + v];

  const int bb = bh / Hc, h = bh % Hc;
#pragma unroll
  for (int dt = 0; dt < 4; dt++)
#pragma unroll
    for (int v = 0; v < 8; v++) {
      const int srow = q0 + v + half * 8;
      const int dcol = h * DKc + dt * 16 + lr;
      Ob[((size_t)(bb * Sc + srow)) * Dc + dcol] = f2bf(o[dt][v] / lf[v]);
    }
}

// ---------------------------------------------------------------------------
// Host launcher
// ---------------------------------------------------------------------------
extern "C" void kernel_launch(void* const* d_in, const int* in_sizes, int n_in,
                              void* d_out, int out_size, void* d_ws, size_t ws_size,
                              hipStream_t stream) {
  const float* query = (const float*)d_in[0];
  const float* key_  = (const float*)d_in[1];
  const float* value = (const float*)d_in[2];
  // d_in[3] is the causal mask; causality is computed analytically instead.
  const float* W_q = (const float*)d_in[4];
  const float* b_q = (const float*)d_in[5];
  const float* W_k = (const float*)d_in[6];
  const float* b_k = (const float*)d_in[7];
  const float* W_v = (const float*)d_in[8];
  const float* b_v = (const float*)d_in[9];
  const float* W_o = (const float*)d_in[10];
  const float* b_o = (const float*)d_in[11];

  char* ws = (char*)d_ws;
  const size_t MB = 1024 * 1024;
  u16* Qb  = (u16*)(ws + 0 * MB);       // [B,H,S,DK] bf16 (pre-scaled by 1/8)
  u16* Kb  = (u16*)(ws + 8 * MB);       // [B,H,S,DK] bf16
  u16* Vt  = (u16*)(ws + 16 * MB);      // [B,H,DK,S] bf16 (transposed)
  u16* Ob  = (u16*)(ws + 24 * MB);      // [B,S,D]    bf16
  u16* Xq  = (u16*)(ws + 32 * MB);      // bf16 copies of activations
  u16* Xk  = (u16*)(ws + 40 * MB);
  u16* Xv  = (u16*)(ws + 48 * MB);
  u16* Wqb = (u16*)(ws + 56 * MB);      // bf16 copies of weights
  u16* Wkb = (u16*)(ws + 58 * MB);
  u16* Wvb = (u16*)(ws + 60 * MB);
  u16* Wob = (u16*)(ws + 62 * MB);

  const dim3 blk(256);
  const int nX = Rc * Dc;               // 4 Mi elements
  const int nW = Dc * Dc;               // 1 Mi elements
  cvt_bf16_kernel<<<nX / 1024, blk, 0, stream>>>(query, Xq, nX);
  cvt_bf16_kernel<<<nX / 1024, blk, 0, stream>>>(key_,  Xk, nX);
  cvt_bf16_kernel<<<nX / 1024, blk, 0, stream>>>(value, Xv, nX);
  cvt_bf16_kernel<<<nW / 1024, blk, 0, stream>>>(W_q, Wqb, nW);
  cvt_bf16_kernel<<<nW / 1024, blk, 0, stream>>>(W_k, Wkb, nW);
  cvt_bf16_kernel<<<nW / 1024, blk, 0, stream>>>(W_v, Wvb, nW);
  cvt_bf16_kernel<<<nW / 1024, blk, 0, stream>>>(W_o, Wob, nW);

  const int projBlocks = (Rc / 16) * (Dc / 64) / 8;     // 512 blocks, 4096 waves
  const int attnBlocks = (Bc * Hc * (Sc / 16)) / 8;     // 512 blocks, 4096 waves

  const float qscale = 0.125f;          // 1/sqrt(DK), folded into Q projection
  proj_kernel<0><<<projBlocks, blk, 0, stream>>>(Xq, Wqb, b_q, Qb, qscale);
  proj_kernel<0><<<projBlocks, blk, 0, stream>>>(Xk, Wkb, b_k, Kb, 1.0f);
  proj_kernel<1><<<projBlocks, blk, 0, stream>>>(Xv, Wvb, b_v, Vt, 1.0f);
  attn_kernel<<<attnBlocks, blk, 0, stream>>>(Qb, Kb, Vt, Ob);
  proj_kernel<2><<<projBlocks, blk, 0, stream>>>(Ob, Wob, b_o, d_out, 1.0f);
}
